// Discriminator_29472065585479
// MI455X (gfx1250) — matrix operand order repaired and checked by reading, not compiled, not run on hardware
//
#include <hip/hip_runtime.h>

// ---------------------------------------------------------------------------
// CDNA5 (gfx1250, wave32) CSLS retrieval: bf16 WMMA GEMMs + fused top-K.
// Round 5: sched_barrier(0) between the B-tile load clause and the WMMA chain
// so the machine scheduler cannot re-serialize load->wait(0)->wmma pairs.
// ---------------------------------------------------------------------------

typedef __attribute__((ext_vector_type(16))) __bf16 v16bf;
typedef __attribute__((ext_vector_type(8)))  __bf16 v8bf;
typedef __attribute__((ext_vector_type(8)))  float  v8f;

#define VSn 20000
#define VTn 20000
#define Bn  2000
#define DSn 300
#define DCn 1024
#define LDP 320   // bf16 feature stride (300 padded to mult of 32, zero tail)
#define LDY 304   // f32 feature stride

// ---------------- WMMA helpers (layouts per CDNA5 ISA 7.12.2) ----------------

__device__ inline v8f wmma_bf16(v16bf a, v16bf b, v8f c) {
  // (neg_a, A, neg_b, B, c_mod, C, reuse_a, reuse_b)
  return __builtin_amdgcn_wmma_f32_16x16x32_bf16(false, a, false, b, (short)0, c,
                                                 false, false);
}

// A-operand 16x32 bf16: lane l holds row (l&15), K = {8h..8h+7} U {16+8h..16+8h+7}
__device__ inline v16bf load_a(const __bf16* __restrict__ A, int lda, int m0,
                               int k0, int lane) {
  int r = lane & 15, h = lane >> 4;
  const __bf16* p = A + (size_t)(m0 + r) * lda + k0 + 8 * h;
  v8bf a0 = *(const v8bf*)p;
  v8bf a1 = *(const v8bf*)(p + 16);
  v16bf a;
#pragma unroll
  for (int i = 0; i < 8; i++) { a[i] = a0[i]; a[8 + i] = a1[i]; }
  return a;
}

__device__ inline v16bf load_a(const float* __restrict__ A, int lda, int m0,
                               int k0, int lane) {
  int r = lane & 15, h = lane >> 4;
  const float* p = A + (size_t)(m0 + r) * lda + k0 + 8 * h;
  const float4* q0 = (const float4*)p;
  const float4* q1 = (const float4*)(p + 16);
  float4 x0 = q0[0], x1 = q0[1], x2 = q1[0], x3 = q1[1];
  v16bf a;
  a[0] = (__bf16)x0.x; a[1] = (__bf16)x0.y; a[2] = (__bf16)x0.z; a[3] = (__bf16)x0.w;
  a[4] = (__bf16)x1.x; a[5] = (__bf16)x1.y; a[6] = (__bf16)x1.z; a[7] = (__bf16)x1.w;
  a[8]  = (__bf16)x2.x; a[9]  = (__bf16)x2.y; a[10] = (__bf16)x2.z; a[11] = (__bf16)x2.w;
  a[12] = (__bf16)x3.x; a[13] = (__bf16)x3.y; a[14] = (__bf16)x3.z; a[15] = (__bf16)x3.w;
  return a;
}

// B-operand 32x16 bf16: lane l holds column (l&15) = matrix row n0+(l&15),
// K = {8h..8h+7} U {16+8h..16+8h+7} (matches A) -> direct row-major loads give X @ Y^T.
__device__ inline v16bf load_b(const __bf16* __restrict__ Wb, int ldw, int n0,
                               int k0, int lane) {
  int c = lane & 15, h = lane >> 4;
  const __bf16* p = Wb + (size_t)(n0 + c) * ldw + k0 + 8 * h;
  v8bf b0 = *(const v8bf*)p;
  v8bf b1 = *(const v8bf*)(p + 16);
  v16bf b;
#pragma unroll
  for (int i = 0; i < 8; i++) { b[i] = b0[i]; b[8 + i] = b1[i]; }
  return b;
}

// load all 10 K-chunks (K=320) of one 16-column B tile
__device__ inline void load_b10(v16bf (&B)[10], const __bf16* __restrict__ P,
                                int n0, int lane) {
#pragma unroll
  for (int ks = 0; ks < 10; ks++) B[ks] = load_b(P, LDP, n0, ks * 32, lane);
}

// ---------------- top-k insertion (sorted descending, register resident) ----

__device__ inline void topk_insert(float (&tv)[10], float v) {
  float cv = v;
#pragma unroll
  for (int i = 0; i < 10; i++) {
    float o = tv[i];
    bool take = cv > o;
    tv[i] = take ? cv : o;
    cv    = take ? o  : cv;
  }
}

__device__ inline void topk_insert_idx(float (&tv)[10], int (&ti)[10], float v,
                                       int id) {
  float cv = v; int cid = id;
#pragma unroll
  for (int i = 0; i < 10; i++) {
    float o = tv[i]; int oi = ti[i];
    bool take = cv > o;
    tv[i] = take ? cv : o;  ti[i] = take ? cid : oi;
    cv    = take ? o  : cv; cid   = take ? oi  : cid;
  }
}

// ---------------- utility kernels -------------------------------------------

__global__ void zero_u32(unsigned* __restrict__ p, long long n) {
  long long i  = (long long)blockIdx.x * blockDim.x + threadIdx.x;
  long long st = (long long)gridDim.x * blockDim.x;
  for (; i < n; i += st) p[i] = 0u;
}

// f32 [rows][cols] -> bf16 [rowsPad][ldw], zero padded
__global__ void conv_w(const float* __restrict__ W, int rows, int cols,
                       __bf16* __restrict__ Wb, int rowsPad, int ldw) {
  long long n  = (long long)rowsPad * ldw;
  long long i  = (long long)blockIdx.x * blockDim.x + threadIdx.x;
  long long st = (long long)gridDim.x * blockDim.x;
  for (; i < n; i += st) {
    int r = (int)(i / ldw), c = (int)(i % ldw);
    float v = (r < rows && c < cols) ? W[(size_t)r * cols + c] : 0.f;
    Wb[i] = (__bf16)v;
  }
}

// wave-per-row L2 normalize in place
__global__ void l2norm_rows(float* __restrict__ X, int ld, int dim, int nrows) {
  int lane = threadIdx.x & 31;
  int row  = blockIdx.x * (blockDim.x >> 5) + (threadIdx.x >> 5);
  if (row >= nrows) return;
  float* p = X + (size_t)row * ld;
  float s = 0.f;
  for (int c = lane; c < dim; c += 32) { float v = p[c]; s += v * v; }
#pragma unroll
  for (int o = 16; o; o >>= 1) s += __shfl_xor(s, o, 32);
  float inv = rsqrtf(fmaxf(s, 1e-30f));
  for (int c = lane; c < dim; c += 32) p[c] *= inv;
}

// column sums over (optionally gathered) rows; threads own disjoint columns
__global__ void colsum_kernel(const float* __restrict__ X, int ld, int dim,
                              const int* __restrict__ idx, int nrows,
                              int rowsPerBlock, float* __restrict__ colsum) {
  int r0 = blockIdx.x * rowsPerBlock;
  int r1 = r0 + rowsPerBlock; if (r1 > nrows) r1 = nrows;
  int t = threadIdx.x;                     // 256 threads
  float s0 = 0.f, s1 = 0.f;
  for (int r = r0; r < r1; r++) {
    int rr = idx ? idx[r] : r;
    const float* p = X + (size_t)rr * ld;
    if (t < dim)       s0 += p[t];
    if (t + 256 < dim) s1 += p[t + 256];
  }
  if (t < dim)       atomicAdd(&colsum[t], s0);
  if (t + 256 < dim) atomicAdd(&colsum[t + 256], s1);
}

// out[r] = l2(stat[gather]) + w * ncn_row, written bf16 into zero-padded buffer
__global__ void combine_kernel(const float* __restrict__ stat, int lds_,
                               const int* __restrict__ statIdx,
                               const float* __restrict__ X, int ldx,
                               const int* __restrict__ xIdx,
                               const float* __restrict__ colsum, float invN,
                               const float* __restrict__ wvec, int nrows,
                               int dim, __bf16* __restrict__ out, int ldo) {
  int lane = threadIdx.x & 31;
  int row  = blockIdx.x * (blockDim.x >> 5) + (threadIdx.x >> 5);
  if (row >= nrows) return;
  const float* sp = stat + (size_t)(statIdx ? statIdx[row] : row) * lds_;
  const float* xp = X    + (size_t)(xIdx    ? xIdx[row]    : row) * ldx;
  float sv[10], xv[10];
  float ss = 0.f, sx = 0.f;
#pragma unroll
  for (int j = 0; j < 10; j++) {
    int c = lane + 32 * j;
    float a = (c < dim) ? sp[c] : 0.f;
    float b = (c < dim) ? (xp[c] - colsum[c] * invN) : 0.f;
    sv[j] = a; xv[j] = b; ss += a * a; sx += b * b;
  }
#pragma unroll
  for (int o = 16; o; o >>= 1) { ss += __shfl_xor(ss, o, 32); sx += __shfl_xor(sx, o, 32); }
  float is = rsqrtf(fmaxf(ss, 1e-30f));
  float ix = rsqrtf(fmaxf(sx, 1e-30f));
#pragma unroll
  for (int j = 0; j < 10; j++) {
    int c = lane + 32 * j;
    if (c < dim) out[(size_t)row * ldo + c] = (__bf16)(sv[j] * is + wvec[c] * xv[j] * ix);
  }
}

// ---------------- WMMA GEMM + bias + tanh -----------------------------------
// out[m][n] = OT(tanh(sum_k A[m][k]*Wb[n][k] + bias[n])), n<Nout else 0.
// grid=(M/16, ceil(Nout/64)), block=128 (4 waves, one 16-col chunk each).
// K loop grouped by G steps: one load clause of 2G tiles, then G chained WMMAs.
template <typename AT, typename OT, int G>
__global__ void mlp_gemm(const AT* __restrict__ A, int lda,
                         const __bf16* __restrict__ Wb, int ldw,
                         const float* __restrict__ bias, int Nout, int Ksteps,
                         OT* __restrict__ out, int ldo) {
  int lane = threadIdx.x & 31;
  int wave = threadIdx.x >> 5;
  int m0 = blockIdx.x * 16;
  int n0 = (blockIdx.y * 4 + wave) * 16;
  int npad = (Nout + 15) & ~15;
  if (n0 >= npad) return;                       // wave-uniform
  v8f acc = {};
#pragma unroll 1
  for (int ks0 = 0; ks0 < Ksteps; ks0 += G) {
    v16bf a[G], b[G];
#pragma unroll
    for (int g = 0; g < G; g++) {
      a[g] = load_a(A, lda, m0, (ks0 + g) * 32, lane);
      b[g] = load_b(Wb, ldw, n0, (ks0 + g) * 32, lane);
    }
    __builtin_amdgcn_sched_barrier(0);   // keep the load clause ahead of WMMAs
#pragma unroll
    for (int g = 0; g < G; g++) acc = wmma_bf16(a[g], b[g], acc);
  }
  int c = lane & 15, h = lane >> 4;
  int n = n0 + c;
  float bv = (n < Nout) ? bias[n] : 0.f;
#pragma unroll
  for (int v = 0; v < 8; v++) {
    int m = m0 + v + 8 * h;
    float z = (n < Nout) ? tanhf(acc[v] + bv) : 0.f;
    out[(size_t)m * ldo + n] = (OT)z;
  }
}

// ---------------- backward CSLS: bwd[t] = mean(top10_s(tgt[t].src[s])) ------
// 1 WG (128 thr) per 16 target rows; 512-col LDS blocks; fused WMMA + top-10.
// B tile loaded as one 20-load clause, then 10 chained WMMAs with staggered
// partial waits (sched_barrier keeps the scheduler from pairing load+wmma).
__global__ void __launch_bounds__(128, 1)
bwd_kernel(const __bf16* __restrict__ Tg,   // [VT][LDP]
           const __bf16* __restrict__ Sr,   // [VS][LDP]
           float* __restrict__ bwd) {
  __shared__ float sc[16 * 516];
  int lane = threadIdx.x & 31;
  int wave = threadIdx.x >> 5;
  int t0 = blockIdx.x * 16;

  v16bf Areg[10];
#pragma unroll
  for (int ks = 0; ks < 10; ks++) Areg[ks] = load_a(Tg, LDP, t0, ks * 32, lane);

  int srow = threadIdx.x & 15;   // scan row
  int ssub = threadIdx.x >> 4;   // 0..7
  float topv[10];
#pragma unroll
  for (int i = 0; i < 10; i++) topv[i] = -1e30f;

  int nblk = (VSn + 511) / 512;
#pragma unroll 1
  for (int blk = 0; blk < nblk; blk++) {
    int c0 = blk * 512;
    int ncols = VSn - c0; if (ncols > 512) ncols = 512;
#pragma unroll 1
    for (int j = 0; j < 8; j++) {
      int cc = (wave * 8 + j) * 16;
      if (cc < ncols) {
        int n0 = c0 + cc;
        // keep next tile's lines in flight while the WMMA chain drains
        int pfc = (j < 7) ? (n0 + 16) : (c0 + 512 + wave * 128);
        if (pfc < VSn) __builtin_prefetch(Sr + (size_t)(pfc + (lane & 15)) * LDP, 0, 1);
        v16bf Breg[10];
        load_b10(Breg, Sr, n0, lane);
        __builtin_amdgcn_sched_barrier(0);
        v8f acc = {};
#pragma unroll
        for (int ks = 0; ks < 10; ks++) acc = wmma_bf16(Areg[ks], Breg[ks], acc);
        int c = lane & 15, h = lane >> 4;
#pragma unroll
        for (int v = 0; v < 8; v++) sc[(v + 8 * h) * 516 + cc + c] = acc[v];
      }
    }
    __syncthreads();
    const float* rowp = &sc[srow * 516];
    int cbase = ssub * 64;
    for (int i = 0; i < 64; i++) {
      int c = cbase + i;
      if (c < ncols) {
        float v = rowp[c];
        if (v > topv[9]) topk_insert(topv, v);
      }
    }
    __syncthreads();
  }
  // merge 8x10 candidates per row
#pragma unroll
  for (int i = 0; i < 10; i++) sc[srow * 516 + ssub * 10 + i] = topv[i];
  __syncthreads();
  if (threadIdx.x < 16) {
    float* cp = &sc[threadIdx.x * 516];
    float sum = 0.f;
    for (int kth = 0; kth < 10; kth++) {
      float best = -3e38f; int bi = 0;
      for (int i = 0; i < 80; i++) { float v = cp[i]; if (v > best) { best = v; bi = i; } }
      sum += best; cp[bi] = -3e38f;
    }
    bwd[t0 + threadIdx.x] = sum * 0.1f;
  }
}

// ---------------- final: top-10 indices of 2*q.tgt - bwd[t] over VT ---------
__global__ void __launch_bounds__(128, 1)
final_topk_kernel(const __bf16* __restrict__ Q,   // [B][LDP]
                  const __bf16* __restrict__ Tg,  // [VT][LDP]
                  const float* __restrict__ bwd,
                  int* __restrict__ outIdx) {
  __shared__ float sc[16 * 516];
  __shared__ int   ci[16 * 80];
  int lane = threadIdx.x & 31;
  int wave = threadIdx.x >> 5;
  int b0 = blockIdx.x * 16;

  v16bf Areg[10];
#pragma unroll
  for (int ks = 0; ks < 10; ks++) Areg[ks] = load_a(Q, LDP, b0, ks * 32, lane);

  int srow = threadIdx.x & 15;
  int ssub = threadIdx.x >> 4;
  float topv[10]; int topi[10];
#pragma unroll
  for (int i = 0; i < 10; i++) { topv[i] = -3e38f; topi[i] = 0; }

  int nblk = (VTn + 511) / 512;
#pragma unroll 1
  for (int blk = 0; blk < nblk; blk++) {
    int c0 = blk * 512;
    int ncols = VTn - c0; if (ncols > 512) ncols = 512;
#pragma unroll 1
    for (int j = 0; j < 8; j++) {
      int cc = (wave * 8 + j) * 16;
      if (cc < ncols) {
        int n0 = c0 + cc;
        int pfc = (j < 7) ? (n0 + 16) : (c0 + 512 + wave * 128);
        if (pfc < VTn) __builtin_prefetch(Tg + (size_t)(pfc + (lane & 15)) * LDP, 0, 1);
        v16bf Breg[10];
        load_b10(Breg, Tg, n0, lane);
        __builtin_amdgcn_sched_barrier(0);
        v8f acc = {};
#pragma unroll
        for (int ks = 0; ks < 10; ks++) acc = wmma_bf16(Areg[ks], Breg[ks], acc);
        int c = lane & 15, h = lane >> 4;
#pragma unroll
        for (int v = 0; v < 8; v++) sc[(v + 8 * h) * 516 + cc + c] = acc[v];
      }
    }
    __syncthreads();
    const float* rowp = &sc[srow * 516];
    int cbase = ssub * 64;
    for (int i = 0; i < 64; i++) {
      int c = cbase + i;
      if (c < ncols) {
        int t = c0 + c;
        float v = 2.f * rowp[c] - bwd[t];
        if (v > topv[9]) topk_insert_idx(topv, topi, v, t);
      }
    }
    __syncthreads();
  }
#pragma unroll
  for (int i = 0; i < 10; i++) {
    sc[srow * 516 + ssub * 10 + i] = topv[i];
    ci[srow * 80 + ssub * 10 + i]  = topi[i];
  }
  __syncthreads();
  if (threadIdx.x < 16) {
    float* cv = &sc[threadIdx.x * 516];
    int*   cj = &ci[threadIdx.x * 80];
    for (int kth = 0; kth < 10; kth++) {
      float best = -3e38f; int bi = 0;
      for (int i = 0; i < 80; i++) { float v = cv[i]; if (v > best) { best = v; bi = i; } }
      outIdx[(size_t)(b0 + threadIdx.x) * 10 + kth] = cj[bi];
      cv[bi] = -3e38f;
    }
  }
}

// ---------------------------------------------------------------------------

extern "C" void kernel_launch(void* const* d_in, const int* in_sizes, int n_in,
                              void* d_out, int out_size, void* d_ws,
                              size_t ws_size, hipStream_t stream) {
  (void)in_sizes; (void)n_in; (void)out_size; (void)ws_size;
  const int*   sid  = (const int*)d_in[0];
  const int*   cid  = (const int*)d_in[1];
  const float* Ssrc = (const float*)d_in[2];
  const float* Stgt = (const float*)d_in[3];
  const float* Csrc = (const float*)d_in[4];
  const float* Ctgt = (const float*)d_in[5];
  const float* W1 = (const float*)d_in[6];  const float* b1 = (const float*)d_in[7];
  const float* W2 = (const float*)d_in[8];  const float* b2 = (const float*)d_in[9];
  const float* W3 = (const float*)d_in[10]; const float* b3 = (const float*)d_in[11];
  const float* W4 = (const float*)d_in[12]; const float* b4 = (const float*)d_in[13];
  const float* w1 = (const float*)d_in[14]; const float* w2 = (const float*)d_in[15];

  char* ws = (char*)d_ws;
  size_t off = 0;
  auto alloc = [&](size_t bytes) -> char* {
    size_t o = (off + 255) & ~(size_t)255;
    off = o + bytes;
    return ws + o;
  };
  __bf16* Hbf  = (__bf16*)alloc((size_t)VSn * LDP * 2);   // MLP intermediate
  __bf16* W1b  = (__bf16*)alloc((size_t)320 * 1024 * 2);
  __bf16* W2b  = (__bf16*)alloc((size_t)320 * 1024 * 2);
  __bf16* W3b  = (__bf16*)alloc((size_t)320 * LDP * 2);
  __bf16* W4b  = (__bf16*)alloc((size_t)320 * LDP * 2);
  float*  Y    = (float*) alloc((size_t)VSn * LDY * 4);   // MLP output (reused)
  float*  csum = (float*) alloc(512 * 4);
  __bf16* SrC  = (__bf16*)alloc((size_t)VSn * LDP * 2);   // src_comb bf16
  __bf16* TgC  = (__bf16*)alloc((size_t)VTn * LDP * 2);   // tgt_comb bf16
  __bf16* QC   = (__bf16*)alloc((size_t)Bn  * LDP * 2);   // query comb bf16
  float*  bwdv = (float*) alloc((size_t)VTn * 4);

  // zero padded bf16 buffers once per launch (padding must be 0 for K loops)
  zero_u32<<<1024, 256, 0, stream>>>((unsigned*)Hbf, (long long)VSn * LDP / 2);
  zero_u32<<<1024, 256, 0, stream>>>((unsigned*)SrC, (long long)VSn * LDP / 2);
  zero_u32<<<1024, 256, 0, stream>>>((unsigned*)TgC, (long long)VTn * LDP / 2);
  zero_u32<<<64,   256, 0, stream>>>((unsigned*)QC,  (long long)Bn  * LDP / 2);

  conv_w<<<512, 256, 0, stream>>>(W1, DSn, DCn, W1b, 320, 1024);
  conv_w<<<512, 256, 0, stream>>>(W2, DSn, DCn, W2b, 320, 1024);
  conv_w<<<256, 256, 0, stream>>>(W3, DSn, DSn, W3b, 320, LDP);
  conv_w<<<256, 256, 0, stream>>>(W4, DSn, DSn, W4b, 320, LDP);

  // ---- source side: Y = tanh(tanh(Csrc@W1^T+b1)@W3^T+b3), then ncn pieces
  mlp_gemm<float,  __bf16, 4><<<dim3(VSn / 16, 5), 128, 0, stream>>>(
      Csrc, DCn, W1b, 1024, b1, DSn, 32, Hbf, LDP);
  mlp_gemm<__bf16, float, 5><<<dim3(VSn / 16, 5), 128, 0, stream>>>(
      Hbf, LDP, W3b, LDP, b3, DSn, 10, Y, LDY);
  l2norm_rows<<<VSn / 8, 256, 0, stream>>>(Y, LDY, DSn, VSn);

  zero_u32<<<1, 256, 0, stream>>>((unsigned*)csum, 512);
  colsum_kernel<<<200, 256, 0, stream>>>(Y, LDY, DSn, nullptr, VSn, 100, csum);
  combine_kernel<<<VSn / 8, 256, 0, stream>>>(Ssrc, DSn, nullptr, Y, LDY, nullptr,
                                              csum, 1.f / VSn, w1, VSn, DSn, SrC, LDP);

  // queries: gather normalized rows of Y via context ids, own batch mean
  zero_u32<<<1, 256, 0, stream>>>((unsigned*)csum, 512);
  colsum_kernel<<<20, 256, 0, stream>>>(Y, LDY, DSn, cid, Bn, 100, csum);
  combine_kernel<<<Bn / 8, 256, 0, stream>>>(Ssrc, DSn, sid, Y, LDY, cid,
                                             csum, 1.f / Bn, w1, Bn, DSn, QC, LDP);

  // ---- target side
  mlp_gemm<float,  __bf16, 4><<<dim3(VTn / 16, 5), 128, 0, stream>>>(
      Ctgt, DCn, W2b, 1024, b2, DSn, 32, Hbf, LDP);
  mlp_gemm<__bf16, float, 5><<<dim3(VTn / 16, 5), 128, 0, stream>>>(
      Hbf, LDP, W4b, LDP, b4, DSn, 10, Y, LDY);
  l2norm_rows<<<VTn / 8, 256, 0, stream>>>(Y, LDY, DSn, VTn);

  zero_u32<<<1, 256, 0, stream>>>((unsigned*)csum, 512);
  colsum_kernel<<<200, 256, 0, stream>>>(Y, LDY, DSn, nullptr, VTn, 100, csum);
  combine_kernel<<<VTn / 8, 256, 0, stream>>>(Stgt, DSn, nullptr, Y, LDY, nullptr,
                                              csum, 1.f / VTn, w2, VTn, DSn, TgC, LDP);

  // ---- backward CSLS term (dominant 240 GFLOP GEMM, operands L2-resident)
  bwd_kernel<<<VTn / 16, 128, 0, stream>>>(TgC, SrC, bwdv);

  // ---- forward sim fused with final top-10 index selection
  final_topk_kernel<<<Bn / 16, 128, 0, stream>>>(QC, TgC, bwdv, (int*)d_out);
}
